// MaskHead_11158325035620
// MI455X (gfx1250) — matrix operand order repaired and checked by reading, not compiled, not run on hardware
//
#include <hip/hip_runtime.h>
#include <hip/hip_bf16.h>
#include <math.h>

// ---------------------------------------------------------------------------
// YOLO mask-head pipeline for MI455X (gfx1250, wave32, WMMA).
//   score -> NMS -> BN-fold -> weight pack (bf16, WMMA-B layout) ->
//   ROI-align (bf16 NHWC) -> implicit-im2col GEMM stack on
//   v_wmma_f32_16x16x32_bf16 (fp32 accum, fused BN + ReLU/sigmoid epilogues).
//   Conv padding handled via a zero guard page (no per-chunk cndmask in the
//   hot loop: OOB taps stream zeros from an L0-resident line).
// Output (float): masks(100,80,28,28) | det_box 400 | det_s 100 |
//                 det_cls 100 | det_valid 100
// ---------------------------------------------------------------------------

typedef __attribute__((ext_vector_type(16))) __bf16 v16bf;
typedef __attribute__((ext_vector_type(8)))  float  v8f;

#define NANCH 16128      // 3*(64^2+32^2+16^2)
#define NB    2
#define NROI  100        // B * MAX_DET
#define CIN   1792
#define NCLS  80

// ---------------------------------------------------------------------------
// 0) zero guard page (re-zeroed every call; harness only poisons ws once)
__global__ void zero_page(__bf16* z) {
  int i = blockIdx.x * 256 + threadIdx.x;
  if (i < CIN) z[i] = (__bf16)0.0f;
}

// ---------------------------------------------------------------------------
// 1) per-anchor score
__global__ void score_kernel(const float* __restrict__ yolo,
                             float* __restrict__ conf, int* __restrict__ cls) {
  int i = blockIdx.x * 256 + threadIdx.x;
  if (i >= NB * NANCH) return;
  const float* p = yolo + (size_t)i * 85;
  float obj = p[4];
  float best = -__builtin_inff(); int bc = 0;
  for (int c = 0; c < NCLS; ++c) {
    float s = p[5 + c] * obj;
    if (s > best) { best = s; bc = c; }
  }
  bool valid = (obj > 0.4f) && (best > 0.4f);
  conf[i] = valid ? best : -__builtin_inff();
  cls[i]  = bc;
}

// ---------------------------------------------------------------------------
// 2) NMS: one block (256 threads) per image
__global__ __launch_bounds__(256)
void nms_kernel(const float* __restrict__ yolo, float* __restrict__ conf,
                const int* __restrict__ cls, float* __restrict__ outTail,
                float* __restrict__ wsBox, float* __restrict__ wsValid) {
  __shared__ float sV[256]; __shared__ int sI[256];
  __shared__ float tS[256]; __shared__ float tB[256][4];
  __shared__ int   tC[256]; __shared__ int live[256]; __shared__ float ks[256];
  const int img = blockIdx.x, tid = threadIdx.x;
  float* cf = conf + (size_t)img * NANCH;
  const float* yo = yolo + (size_t)img * NANCH * 85;
  const float NEG = -__builtin_inff();

  // top-256 (destructive argmax rounds; tie-break = lowest index like top_k)
  for (int it = 0; it < 256; ++it) {
    float bv = NEG; int bi = NANCH;
    for (int j = tid; j < NANCH; j += 256) {
      float v = cf[j];
      if (v > bv || (v == bv && j < bi)) { bv = v; bi = j; }
    }
    sV[tid] = bv; sI[tid] = bi;
    for (int s = 128; s > 0; s >>= 1) {
      __syncthreads();
      if (tid < s) {
        if (sV[tid+s] > sV[tid] || (sV[tid+s] == sV[tid] && sI[tid+s] < sI[tid]))
        { sV[tid] = sV[tid+s]; sI[tid] = sI[tid+s]; }
      }
    }
    __syncthreads();
    if (tid == 0) {
      int w = sI[0]; float v = sV[0];
      tS[it] = v;
      int wc = (w < NANCH) ? w : 0;
      const float* pb = yo + (size_t)wc * 85;
      float cx = pb[0], cy = pb[1], ww = pb[2], hh = pb[3];
      tB[it][0] = cx - ww*0.5f; tB[it][1] = cy - hh*0.5f;
      tB[it][2] = cx + ww*0.5f; tB[it][3] = cy + hh*0.5f;
      tC[it] = cls[(size_t)img*NANCH + wc];
      if (w < NANCH) cf[w] = NEG;
    }
    __syncthreads();
  }

  // sequential suppression (class-offset IoU)
  live[tid] = (tS[tid] > NEG) ? 1 : 0;
  __syncthreads();
  float offJ = (float)tC[tid] * 4096.0f;
  float jx0 = tB[tid][0]+offJ, jy0 = tB[tid][1]+offJ;
  float jx1 = tB[tid][2]+offJ, jy1 = tB[tid][3]+offJ;
  float areaJ = (jx1-jx0)*(jy1-jy0);
  for (int i = 0; i < 256; ++i) {
    bool li = (live[i] != 0);
    if (li && tid > i) {
      float offI = (float)tC[i]*4096.0f;
      float ix0 = tB[i][0]+offI, iy0 = tB[i][1]+offI;
      float ix1 = tB[i][2]+offI, iy1 = tB[i][3]+offI;
      float areaI = (ix1-ix0)*(iy1-iy0);
      float lx = fmaxf(ix0, jx0), ly = fmaxf(iy0, jy0);
      float rx = fminf(ix1, jx1), ry = fminf(iy1, jy1);
      float w = fmaxf(rx-lx, 0.f), h = fmaxf(ry-ly, 0.f);
      float inter = w*h;
      float iou = inter / (areaI + areaJ - inter + 1e-9f);
      if (iou > 0.5f) live[tid] = 0;
    }
    __syncthreads();
  }
  ks[tid] = live[tid] ? tS[tid] : NEG;
  __syncthreads();

  // top-50
  for (int d = 0; d < 50; ++d) {
    sV[tid] = ks[tid]; sI[tid] = tid;
    for (int s = 128; s > 0; s >>= 1) {
      __syncthreads();
      if (tid < s) {
        if (sV[tid+s] > sV[tid] || (sV[tid+s] == sV[tid] && sI[tid+s] < sI[tid]))
        { sV[tid] = sV[tid+s]; sI[tid] = sI[tid+s]; }
      }
    }
    __syncthreads();
    if (tid == 0) {
      int w = sI[0]; float v = sV[0];
      float vm = (v > NEG) ? 1.f : 0.f;
      int r = img*50 + d;
      for (int k = 0; k < 4; ++k) {
        float b = tB[w][k] * vm;
        outTail[r*4 + k] = b;
        wsBox[r*4 + k]   = b;
      }
      outTail[400 + r] = (vm > 0.f) ? v : 0.f;
      outTail[500 + r] = (vm > 0.f) ? (float)tC[w] : 0.f;
      outTail[600 + r] = vm;
      wsValid[r] = vm;
      ks[w] = NEG;
    }
    __syncthreads();
  }
}

// ---------------------------------------------------------------------------
// 3) fold BN into scale/bias
__global__ void bn_fuse(const float* g, const float* b, const float* m,
                        const float* v, int n, float* sc, float* bi) {
  int i = blockIdx.x * blockDim.x + threadIdx.x;
  if (i < n) {
    float s = g[i] * rsqrtf(v[i] + 1e-5f);
    sc[i] = s; bi[i] = b[i] - m[i]*s;
  }
}

// ---------------------------------------------------------------------------
// 4) weight packing -> bf16 [k/32][N][32]  (WMMA B fragment friendly)
__global__ void pack_conv3(const float* __restrict__ w, __bf16* __restrict__ out,
                           int Cin, int N, int K) {
  int idx = blockIdx.x * 256 + threadIdx.x;
  if (idx >= K * N) return;
  int kb = idx / (N*32); int rem = idx % (N*32); int n = rem/32; int kk = rem%32;
  int k = kb*32 + kk; int kpos = k / Cin; int c = k % Cin;
  int ky = kpos/3, kx = kpos%3;
  out[idx] = (__bf16)w[((size_t)(n*Cin + c)*3 + ky)*3 + kx];
}

__global__ void pack_deconv(const float* __restrict__ wt4, __bf16* __restrict__ out) {
  // out: [parity p=py*2+px][k/32=8][256][32]; weight = wt4[ci][co][1-py][1-px]
  int idx = blockIdx.x * 256 + threadIdx.x;
  if (idx >= 4*256*256) return;
  int p = idx / (256*256); int rem = idx % (256*256);
  int kb = rem / (256*32); int r2 = rem % (256*32); int n = r2/32; int kk = r2%32;
  int ci = kb*32 + kk; int py = p >> 1, px = p & 1;
  out[idx] = (__bf16)wt4[((size_t)(ci*256 + n)*2 + (1-py))*2 + (1-px)];
}

__global__ void pack_w5(const float* __restrict__ w5, __bf16* __restrict__ out) {
  int idx = blockIdx.x * 256 + threadIdx.x;
  if (idx >= 256*NCLS) return;
  int kb = idx / (NCLS*32); int rem = idx % (NCLS*32); int n = rem/32; int kk = rem%32;
  int c = kb*32 + kk;
  out[idx] = (__bf16)w5[n*256 + c];
}

// ---------------------------------------------------------------------------
// 5) ROI-Align into bf16 NHWC activation buffer
__global__ __launch_bounds__(256)
void roi_align(const float* __restrict__ feat, const float* __restrict__ boxes,
               const float* __restrict__ valid, __bf16* __restrict__ act0,
               int C, int H, float scale, int chOff) {
  const int r = blockIdx.x;           // roi
  const int q = blockIdx.y;           // 0..195 output pixel
  const int py = q / 14, px = q % 14;
  const int img = r / 50;
  const float vflag = valid[r];
  const float* bx = boxes + r*4;
  float x0 = bx[0]*scale - 0.5f, y0 = bx[1]*scale - 0.5f;
  float x1 = bx[2]*scale - 0.5f, y1 = bx[3]*scale - 0.5f;
  float bw = (x1-x0)*(1.0f/14.0f), bh = (y1-y0)*(1.0f/14.0f);
  int   idxs[4][4]; float wts[4][4];
  for (int sy = 0; sy < 2; ++sy)
    for (int sx = 0; sx < 2; ++sx) {
      float gy = ((float)(py*2+sy) + 0.5f) * 0.5f;
      float gx = ((float)(px*2+sx) + 0.5f) * 0.5f;
      float Y = y0 + gy*bh, X = x0 + gx*bw;
      bool v = (Y > -1.f) && (Y < (float)H) && (X > -1.f) && (X < (float)H);
      float y = fminf(fmaxf(Y, 0.f), (float)(H-1));
      float x = fminf(fmaxf(X, 0.f), (float)(H-1));
      int yi = (int)floorf(y), xi = (int)floorf(x);
      int y1i = min(yi+1, H-1), x1i = min(xi+1, H-1);
      float ly = y - (float)yi, lx = x - (float)xi;
      float hy = 1.f - ly, hx = 1.f - lx;
      int s = sy*2 + sx;
      idxs[s][0] = yi*H + xi;  idxs[s][1] = yi*H + x1i;
      idxs[s][2] = y1i*H + xi; idxs[s][3] = y1i*H + x1i;
      float m = v ? 0.25f : 0.f;          // /4 sample average folded in
      wts[s][0] = hy*hx*m; wts[s][1] = hy*lx*m;
      wts[s][2] = ly*hx*m; wts[s][3] = ly*lx*m;
    }
  size_t fbase = (size_t)img * C * H * H;
  size_t obase = ((size_t)r*196 + q) * CIN + chOff;
  for (int c = threadIdx.x; c < C; c += 256) {
    const float* f = feat + fbase + (size_t)c * H * H;
    float acc = 0.f;
    #pragma unroll
    for (int s = 0; s < 4; ++s)
      acc += wts[s][0]*f[idxs[s][0]] + wts[s][1]*f[idxs[s][1]]
           + wts[s][2]*f[idxs[s][2]] + wts[s][3]*f[idxs[s][3]];
    act0[obase + c] = (__bf16)(acc * vflag);
  }
}

// ---------------------------------------------------------------------------
// 6) implicit-im2col GEMM on v_wmma_f32_16x16x32_bf16
//    MODE 0: 3x3 pad1, 14x14, k=(ky*3+kx)*Cin+c -> bf16 NHWC out  (NT=2)
//    MODE 1: deconv parity GEMM -> 26x26 NHWC out                 (NT=2)
//    MODE 2: 1x1 pad1 over 26x26 -> 28x28 sigmoid fp32 NCHW masks (NT=1)
struct GemmP {
  const __bf16* A;  const __bf16* Wp;  const __bf16* Z;   // Z = zero page
  const float*  scale; const float* bias;
  __bf16* Obf;  float* Ofp;
  int M, N, K, Cin, py, px;
};

union Frag { uint4 u[2]; v16bf v; };

__device__ __forceinline__ void ldA(Frag& f, const __bf16* ap, int cc) {
  const __bf16* q = ap + (size_t)cc * 32;
  f.u[0] = *(const uint4*)(q);          // K {0..7}/{8..15}   (by half)
  f.u[1] = *(const uint4*)(q + 16);     // K {16..23}/{24..31}
}

template<int NT>
__device__ __forceinline__ void ldB(Frag (&f)[NT], const __bf16* const (&bp)[NT],
                                    int cc, size_t bStride) {
  #pragma unroll
  for (int nt = 0; nt < NT; ++nt) {
    const __bf16* q = bp[nt] + (size_t)cc * bStride;
    f[nt].u[0] = *(const uint4*)(q);      // B: K {0..7}/{8..15} then {16..23}/{24..31} (by half)
    f[nt].u[1] = *(const uint4*)(q + 16);
  }
}

template<int NT>
__device__ __forceinline__ void mma(v8f (&acc)[NT], const Frag& a,
                                    const Frag (&b)[NT]) {
  #pragma unroll
  for (int nt = 0; nt < NT; ++nt)
    acc[nt] = __builtin_amdgcn_wmma_f32_16x16x32_bf16(
        false, a.v, false, b[nt].v, (short)0, acc[nt], false, false);
}

template<int MODE, int NT>
__global__ __launch_bounds__(256)
void gemm_wmma(GemmP p) {
  const int lane = threadIdx.x & 31;
  const int wave = threadIdx.x >> 5;
  const int half = lane >> 4;            // K-half of the wave (ISA A/B layout)
  const int lidx = lane & 15;
  const int m0 = (blockIdx.x * 8 + wave) * 16;
  const int n0 = blockIdx.y * (16 * NT);
  if (m0 >= p.M) return;
  const int mrow = m0 + lidx;

  const __bf16* bp[NT];
  #pragma unroll
  for (int nt = 0; nt < NT; ++nt)
    bp[nt] = p.Wp + (size_t)(n0 + nt*16 + lidx) * 32 + half * 8;
  const size_t bStride = (size_t)p.N * 32;

  v8f acc[NT];
  #pragma unroll
  for (int nt = 0; nt < NT; ++nt) acc[nt] = (v8f){0.f,0.f,0.f,0.f,0.f,0.f,0.f,0.f};

  Frag a0, a1, b0[NT], b1[NT];
  const __bf16* zp = p.Z + half * 8;     // zero guard page (pads -> zeros)

  if constexpr (MODE == 0) {
    // taps outer (unrolled), channel chunks inner (double-buffered)
    int roi = mrow / 196; int pix = mrow % 196;
    const int oy = pix / 14, ox = pix % 14;
    const __bf16* roiBase = p.A + (size_t)roi * 196 * p.Cin + half * 8;
    const int nIter = p.Cin >> 5;        // 56 (conv1) / 8 (conv2,3) — even
    int kChunk = 0;
    #pragma unroll
    for (int t = 0; t < 9; ++t) {
      const int ky = t / 3, kx = t % 3;  // compile-time
      const int iy = oy + ky - 1, ix = ox + kx - 1;
      const bool ok = ((unsigned)iy < 14u) && ((unsigned)ix < 14u);
      const __bf16* ap = ok ? roiBase + (size_t)(iy*14 + ix) * p.Cin : zp;
      ldA(a0, ap, 0);
      ldB<NT>(b0, bp, kChunk, bStride);
      for (int cc = 0; cc < nIter; cc += 2) {
        __builtin_prefetch((const void*)(ap + (size_t)(cc + 8) * 32), 0, 1);
        ldA(a1, ap, cc + 1);
        ldB<NT>(b1, bp, kChunk + cc + 1, bStride);
        mma<NT>(acc, a0, b0);
        if (cc + 2 < nIter) {
          ldA(a0, ap, cc + 2);
          ldB<NT>(b0, bp, kChunk + cc + 2, bStride);
        }
        mma<NT>(acc, a1, b1);
      }
      kChunk += nIter;
    }
  } else {
    // single fixed input pixel per row (deconv parity / 1x1 pad1)
    const __bf16* apBase = p.Z;          // OOB rows/pixels read zeros
    if constexpr (MODE == 1) {
      if (mrow < p.M) {
        int roi = mrow / 169; int j = mrow % 169; int jy = j/13, jx = j%13;
        int iy = jy + p.py, ix = jx + p.px;        // always in [0,14)
        apBase = p.A + ((size_t)(roi*196 + iy*14 + ix)) * p.Cin;
      }
    } else {
      int roi = mrow / 784; int pix = mrow % 784;
      int iy = pix/28 - 1, ix = pix%28 - 1;
      if (((unsigned)iy < 26u) && ((unsigned)ix < 26u))
        apBase = p.A + ((size_t)(roi*676 + iy*26 + ix)) * p.Cin;
    }
    const __bf16* ap = apBase + half * 8;
    const int nIter = p.K >> 5;                    // 8 — even
    ldA(a0, ap, 0);
    ldB<NT>(b0, bp, 0, bStride);
    for (int cc = 0; cc < nIter; cc += 2) {
      ldA(a1, ap, cc + 1);
      ldB<NT>(b1, bp, cc + 1, bStride);
      mma<NT>(acc, a0, b0);
      if (cc + 2 < nIter) {
        ldA(a0, ap, cc + 2);
        ldB<NT>(b0, bp, cc + 2, bStride);
      }
      mma<NT>(acc, a1, b1);
    }
  }

  // epilogue: C tile element (vr, lane) -> M = m0+vr+8*half, N = n0+nt*16+lidx
  #pragma unroll
  for (int nt = 0; nt < NT; ++nt) {
    const int n = n0 + nt*16 + lidx;
    #pragma unroll
    for (int vr = 0; vr < 8; ++vr) {
      int mr = m0 + vr + (half << 3);
      if (mr >= p.M) continue;
      float x = acc[nt][vr] * p.scale[n] + p.bias[n];
      if constexpr (MODE == 2) {
        x = 1.0f / (1.0f + __expf(-x));           // BN + sigmoid
        int r2 = mr / 784; int pix = mr % 784;
        p.Ofp[((size_t)(r2*NCLS + n)) * 784 + pix] = x;
      } else {
        x = fmaxf(x, 0.f);                         // BN + ReLU
        size_t opix;
        if constexpr (MODE == 0) opix = (size_t)mr;
        else {
          int r2 = mr / 169; int j = mr % 169; int jy = j/13, jx = j%13;
          opix = (size_t)(r2*676 + (2*jy + p.py)*26 + (2*jx + p.px));
        }
        p.Obf[opix * (size_t)p.N + n] = (__bf16)x;
      }
    }
  }
}

// ---------------------------------------------------------------------------
extern "C" void kernel_launch(void* const* d_in, const int* in_sizes, int n_in,
                              void* d_out, int out_size, void* d_ws, size_t ws_size,
                              hipStream_t stream) {
  (void)in_sizes; (void)n_in; (void)out_size; (void)ws_size;
  // setup_inputs() dict order, params flattened in insertion order:
  const float* yolo   = (const float*)d_in[0];
  const float* feat8  = (const float*)d_in[1];
  const float* feat16 = (const float*)d_in[2];
  const float* feat32 = (const float*)d_in[3];
  const float* w1     = (const float*)d_in[4];
  const float* bn1[4] = {(const float*)d_in[5],(const float*)d_in[6],(const float*)d_in[7],(const float*)d_in[8]};
  const float* w2     = (const float*)d_in[9];
  const float* bn2[4] = {(const float*)d_in[10],(const float*)d_in[11],(const float*)d_in[12],(const float*)d_in[13]};
  const float* w3     = (const float*)d_in[14];
  const float* bn3[4] = {(const float*)d_in[15],(const float*)d_in[16],(const float*)d_in[17],(const float*)d_in[18]};
  const float* wt4    = (const float*)d_in[19];
  const float* bn4[4] = {(const float*)d_in[20],(const float*)d_in[21],(const float*)d_in[22],(const float*)d_in[23]};
  const float* w5     = (const float*)d_in[24];
  const float* bn5[4] = {(const float*)d_in[25],(const float*)d_in[26],(const float*)d_in[27],(const float*)d_in[28]};

  float* out = (float*)d_out;
  float* outTail = out + (size_t)NROI * NCLS * 784;   // after masks (6,272,000)

  // ---- workspace carve-up
  char* w = (char*)d_ws;
  auto alloc = [&](size_t bytes) { char* q = w; w += (bytes + 255) & ~(size_t)255; return q; };
  float*  wsConf  = (float*)alloc((size_t)NB*NANCH*4);
  int*    wsCls   = (int*)  alloc((size_t)NB*NANCH*4);
  float*  wsBox   = (float*)alloc((size_t)NROI*4*4);
  float*  wsValid = (float*)alloc((size_t)NROI*4);
  float*  bnS[5], *bnB[5];
  for (int i = 0; i < 5; ++i) { bnS[i] = (float*)alloc(256*4); bnB[i] = (float*)alloc(256*4); }
  __bf16* zpage = (__bf16*)alloc((size_t)CIN*2);        // zero guard page
  __bf16* wp1  = (__bf16*)alloc((size_t)16128*256*2);
  __bf16* wp2  = (__bf16*)alloc((size_t)2304*256*2);
  __bf16* wp3  = (__bf16*)alloc((size_t)2304*256*2);
  __bf16* wp4  = (__bf16*)alloc((size_t)4*256*256*2);
  __bf16* wp5  = (__bf16*)alloc((size_t)256*NCLS*2);
  __bf16* act0 = (__bf16*)alloc((size_t)19600*CIN*2);   // (100,14,14,1792) NHWC
  __bf16* act1 = (__bf16*)alloc((size_t)19600*256*2);
  __bf16* act2 = (__bf16*)alloc((size_t)19600*256*2);
  __bf16* act4 = (__bf16*)alloc((size_t)67600*256*2);   // (100,26,26,256) NHWC

  // ---- 0) zero guard page, 1) scores, 2) NMS
  zero_page<<<(CIN + 255)/256, 256, 0, stream>>>(zpage);
  score_kernel<<<(NB*NANCH + 255)/256, 256, 0, stream>>>(yolo, wsConf, wsCls);
  nms_kernel<<<NB, 256, 0, stream>>>(yolo, wsConf, wsCls, outTail, wsBox, wsValid);

  // ---- 3) BN folding
  bn_fuse<<<1, 256, 0, stream>>>(bn1[0], bn1[1], bn1[2], bn1[3], 256, bnS[0], bnB[0]);
  bn_fuse<<<1, 256, 0, stream>>>(bn2[0], bn2[1], bn2[2], bn2[3], 256, bnS[1], bnB[1]);
  bn_fuse<<<1, 256, 0, stream>>>(bn3[0], bn3[1], bn3[2], bn3[3], 256, bnS[2], bnB[2]);
  bn_fuse<<<1, 256, 0, stream>>>(bn4[0], bn4[1], bn4[2], bn4[3], 256, bnS[3], bnB[3]);
  bn_fuse<<<1, 256, 0, stream>>>(bn5[0], bn5[1], bn5[2], bn5[3], NCLS, bnS[4], bnB[4]);

  // ---- 4) weight packing
  pack_conv3<<<(16128*256 + 255)/256, 256, 0, stream>>>(w1, wp1, CIN, 256, 16128);
  pack_conv3<<<(2304*256  + 255)/256, 256, 0, stream>>>(w2, wp2, 256, 256, 2304);
  pack_conv3<<<(2304*256  + 255)/256, 256, 0, stream>>>(w3, wp3, 256, 256, 2304);
  pack_deconv<<<(4*256*256 + 255)/256, 256, 0, stream>>>(wt4, wp4);
  pack_w5<<<(256*NCLS + 255)/256, 256, 0, stream>>>(w5, wp5);

  // ---- 5) ROI align (3 FPN levels -> NHWC bf16)
  dim3 rg(NROI, 196);
  roi_align<<<rg, 256, 0, stream>>>(feat8,  wsBox, wsValid, act0, 256,  64, 1.f/8.f,  0);
  roi_align<<<rg, 256, 0, stream>>>(feat16, wsBox, wsValid, act0, 512,  32, 1.f/16.f, 256);
  roi_align<<<rg, 256, 0, stream>>>(feat32, wsBox, wsValid, act0, 1024, 16, 1.f/32.f, 768);

  // ---- 6) WMMA GEMM stack
  GemmP g{};
  // conv1: M=19600 N=256 K=16128 (NT=2 -> 32 cols per wave, grid.y=8)
  g = {act0, wp1, zpage, bnS[0], bnB[0], act1, nullptr, 19600, 256, 16128, CIN, 0, 0};
  gemm_wmma<0,2><<<dim3((19600/16 + 7)/8, 8), 256, 0, stream>>>(g);
  // conv2
  g = {act1, wp2, zpage, bnS[1], bnB[1], act2, nullptr, 19600, 256, 2304, 256, 0, 0};
  gemm_wmma<0,2><<<dim3((19600/16 + 7)/8, 8), 256, 0, stream>>>(g);
  // conv3
  g = {act2, wp3, zpage, bnS[2], bnB[2], act1, nullptr, 19600, 256, 2304, 256, 0, 0};
  gemm_wmma<0,2><<<dim3((19600/16 + 7)/8, 8), 256, 0, stream>>>(g);
  // deconv: 4 parity GEMMs, M=100*13*13=16900, out 26x26
  for (int par = 0; par < 4; ++par) {
    g = {act1, wp4 + (size_t)par*256*256, zpage, bnS[3], bnB[3], act4, nullptr,
         16900, 256, 256, 256, par >> 1, par & 1};
    gemm_wmma<1,2><<<dim3(((16900+15)/16 + 7)/8, 8), 256, 0, stream>>>(g);
  }
  // conv5: 1x1 pad1 -> 28x28 masks, sigmoid, fp32 NCHW into d_out
  g = {act4, wp5, zpage, bnS[4], bnB[4], nullptr, out, 78400, NCLS, 256, 256, 0, 0};
  gemm_wmma<2,1><<<dim3((78400/16 + 7)/8, NCLS/16), 256, 0, stream>>>(g);
}
